// PolarAttention_67774583931096
// MI455X (gfx1250) — matrix operand order repaired and checked by reading, not compiled, not run on hardware
//
#include <hip/hip_runtime.h>
#include <math.h>

typedef __bf16 bf16_t;
typedef __attribute__((ext_vector_type(16))) __bf16 v16bf;
typedef __attribute__((ext_vector_type(8)))  __bf16 v8bf;
typedef __attribute__((ext_vector_type(8)))  float  v8f;

#define NEG_INF (-3.0e38f)

static __device__ __forceinline__ v16bf cat16(v8bf lo, v8bf hi) {
  return __builtin_shufflevector(lo, hi, 0,1,2,3,4,5,6,7,8,9,10,11,12,13,14,15);
}

// A fragment: 16x32 bf16 tile, row-major memory with leading dim ld.
// ISA layout: lanes 0-15 (M=lane) hold K{0..7,16..23}; lanes 16-31 (M=lane-16) hold K{8..15,24..31}.
static __device__ __forceinline__ v16bf load_a_frag(const bf16_t* base, int ld) {
  int lane = threadIdx.x & 31;
  int half = lane >> 4;
  int m    = lane & 15;
  const bf16_t* p = base + (size_t)m * ld + (half ? 8 : 0);
  v8bf lo = *(const v8bf*)(p);
  v8bf hi = *(const v8bf*)(p + 16);
  return cat16(lo, hi);
}

// B fragment: 32x16 bf16. Memory layout: element [k][n] at base + n*ld + k
// (i.e., each output column's k-run is contiguous).
// ISA layout: lanes 0-15 (N=lane) hold K{0..7,16..23}; lanes 16-31 (N=lane-16) hold K{8..15,24..31}.
static __device__ __forceinline__ v16bf load_b_frag(const bf16_t* base, int ld) {
  int lane = threadIdx.x & 31;
  int half = lane >> 4;
  int n    = lane & 15;
  const bf16_t* p = base + (size_t)n * ld + (half ? 8 : 0);
  v8bf lo = *(const v8bf*)(p);
  v8bf hi = *(const v8bf*)(p + 16);
  return cat16(lo, hi);
}

static __device__ __forceinline__ v8f wmma_bf16(v16bf a, v16bf b, v8f c) {
  return __builtin_amdgcn_wmma_f32_16x16x32_bf16(false, a, false, b, (short)0, c, false, false);
}

// ---------------------------------------------------------------------------
// f32 -> bf16 conversion
__global__ void f32_to_bf16_kernel(const float* __restrict__ src,
                                   bf16_t* __restrict__ dst, size_t n) {
  size_t i = (size_t)blockIdx.x * blockDim.x + threadIdx.x;
  if (i < n) dst[i] = (bf16_t)src[i];
}

// ---------------------------------------------------------------------------
// Generic WMMA GEMM: C[i,j] = sum_d A[i,d] * Bw[j,d]   (i.e. A @ Bw^T)
// A: [M,K] bf16 row-major; Bw: [Nout,K] bf16 row-major.
// grid = (Nout/128, M/64), block = 128 (4 waves); wave w owns a 16x128 C tile.
template<bool SILU, bool OUT_BF16>
__global__ __launch_bounds__(128, 1)
void gemm_kernel(const bf16_t* __restrict__ A,
                 const bf16_t* __restrict__ Bw,
                 void* __restrict__ out,
                 int M, int Nout, int K) {
  int wave = threadIdx.x >> 5;
  int lane = threadIdx.x & 31;
  int half = lane >> 4;
  int n    = lane & 15;
  int row0 = blockIdx.y * 64 + wave * 16;
  int col0 = blockIdx.x * 128;

  v8f acc[8] = {};
  for (int kb = 0; kb < K; kb += 32) {
    const bf16_t* arow = A + (size_t)row0 * K + kb;
    if (kb + 64 < K)
      __builtin_prefetch(arow + 64, 0, 1);   // global_prefetch_b8 next chunk
    v16bf a = load_a_frag(arow, K);
#pragma unroll
    for (int t = 0; t < 8; ++t) {
      v16bf b = load_b_frag(Bw + (size_t)(col0 + t * 16) * K + kb, K);
      acc[t] = wmma_bf16(a, b, acc[t]);
    }
  }
#pragma unroll
  for (int t = 0; t < 8; ++t) {
#pragma unroll
    for (int r = 0; r < 8; ++r) {
      int row = row0 + r + half * 8;
      int col = col0 + t * 16 + n;
      float v = acc[t][r];
      if (SILU) v = v / (1.0f + __expf(-v));
      if (OUT_BF16) ((bf16_t*)out)[(size_t)row * Nout + col] = (bf16_t)v;
      else          ((float*)out)[(size_t)row * Nout + col]  = v;
    }
  }
}

// ---------------------------------------------------------------------------
// Transpose V: v [B,N,H*64] -> vt [B,H,64,N]  (bf16)
__global__ void transpose_v_kernel(const bf16_t* __restrict__ v,
                                   bf16_t* __restrict__ vt) {
  size_t i = (size_t)blockIdx.x * blockDim.x + threadIdx.x;
  if (i >= (size_t)4 * 16 * 64 * 2048) return;
  int tok = (int)(i & 2047);
  int c   = (int)((i >> 11) & 63);
  int h   = (int)((i >> 17) & 15);
  int b   = (int)(i >> 21);
  vt[i] = v[((size_t)(b * 2048 + tok)) * 1024 + h * 64 + c];
}

// ---------------------------------------------------------------------------
// Scale factors: f[b,h,tok] = r*scale[h]/(r+eps), r = ||row||_2 over hd=64
__global__ void factors_kernel(const bf16_t* __restrict__ q,
                               const float* __restrict__ scale,
                               float* __restrict__ f, float eps) {
  int i = blockIdx.x * blockDim.x + threadIdx.x;
  if (i >= 4 * 16 * 2048) return;
  int tok = i & 2047;
  int h   = (i >> 11) & 15;
  int b   = i >> 15;
  const bf16_t* p = q + ((size_t)(b * 2048 + tok)) * 1024 + h * 64;
  float s = 0.0f;
#pragma unroll 8
  for (int c = 0; c < 64; ++c) { float x = (float)p[c]; s += x * x; }
  float r = sqrtf(s);
  f[i] = r * scale[h] / (r + eps);
}

// ---------------------------------------------------------------------------
// Flash-style causal attention per (b,h). One wave owns 16 query rows.
// grid = (N/64, B*H), block = 128 (4 waves).
__global__ __launch_bounds__(128, 1)
void attn_kernel(const bf16_t* __restrict__ qb,
                 const bf16_t* __restrict__ kb,
                 const bf16_t* __restrict__ vt,
                 const float*  __restrict__ fqv,
                 const float*  __restrict__ fkv,
                 bf16_t* __restrict__ attn) {
  __shared__ __align__(16) bf16_t pbuf[4][16 * 32];
  int wave = threadIdx.x >> 5;
  int lane = threadIdx.x & 31;
  int half = lane >> 4;
  int n    = lane & 15;
  int bh   = blockIdx.y;
  int b    = bh >> 4, h = bh & 15;
  int i0   = blockIdx.x * 64 + wave * 16;

  const bf16_t* qhead = qb + ((size_t)b * 2048) * 1024 + h * 64;
  const bf16_t* khead = kb + ((size_t)b * 2048) * 1024 + h * 64;
  const bf16_t* vth   = vt + (size_t)bh * 64 * 2048;     // [64][2048]
  const float*  fq    = fqv + (size_t)bh * 2048;
  const float*  fk    = fkv + (size_t)bh * 2048;

  // Q fragments (16 rows x hd=64 -> two k=32 chunks), reused for all keys.
  v16bf aq0 = load_a_frag(qhead + (size_t)i0 * 1024 + 0,  1024);
  v16bf aq1 = load_a_frag(qhead + (size_t)i0 * 1024 + 32, 1024);

  float fqr[8];
#pragma unroll
  for (int r = 0; r < 8; ++r) fqr[r] = fq[i0 + r + half * 8];

  v8f o[4] = {};
  float mrow[8], lrow[8];
#pragma unroll
  for (int r = 0; r < 8; ++r) { mrow[r] = NEG_INF; lrow[r] = 0.0f; }

  bf16_t* pb = pbuf[wave];

  for (int jb = 0; jb < i0 + 16; jb += 32) {
    // Scores for keys [jb,jb+16) and [jb+16,jb+32)
    v8f s0 = {}, s1 = {};
    {
      v16bf b00 = load_b_frag(khead + (size_t)jb * 1024 + 0,  1024);
      v16bf b01 = load_b_frag(khead + (size_t)jb * 1024 + 32, 1024);
      s0 = wmma_bf16(aq0, b00, s0);
      s0 = wmma_bf16(aq1, b01, s0);
      v16bf b10 = load_b_frag(khead + (size_t)(jb + 16) * 1024 + 0,  1024);
      v16bf b11 = load_b_frag(khead + (size_t)(jb + 16) * 1024 + 32, 1024);
      s1 = wmma_bf16(aq0, b10, s1);
      s1 = wmma_bf16(aq1, b11, s1);
    }
    float fk0 = fk[jb + n];
    float fk1 = fk[jb + 16 + n];
    float p0[8], p1[8], alpha[8];
#pragma unroll
    for (int r = 0; r < 8; ++r) {
      int row = i0 + r + half * 8;
      float x0 = s0[r] * fqr[r] * fk0;
      float x1 = s1[r] * fqr[r] * fk1;
      if (jb + n > row)      x0 = NEG_INF;
      if (jb + 16 + n > row) x1 = NEG_INF;
      p0[r] = x0; p1[r] = x1;
    }
    // Online softmax: each row lives in 16 lanes of one half-wave.
#pragma unroll
    for (int r = 0; r < 8; ++r) {
      float mloc = fmaxf(p0[r], p1[r]);
#pragma unroll
      for (int sft = 8; sft >= 1; sft >>= 1)
        mloc = fmaxf(mloc, __shfl_xor(mloc, sft, 32));
      float mnew = fmaxf(mrow[r], mloc);
      alpha[r] = __expf(mrow[r] - mnew);
      mrow[r]  = mnew;
      float e0 = __expf(p0[r] - mnew);
      float e1 = __expf(p1[r] - mnew);
      p0[r] = e0; p1[r] = e1;
      float ls = e0 + e1;
#pragma unroll
      for (int sft = 8; sft >= 1; sft >>= 1)
        ls += __shfl_xor(ls, sft, 32);
      lrow[r] = lrow[r] * alpha[r] + ls;
    }
    // Rescale accumulator; stage P (16x32) in LDS to re-layout C->A fragment.
#pragma unroll
    for (int r = 0; r < 8; ++r) {
      o[0][r] *= alpha[r]; o[1][r] *= alpha[r];
      o[2][r] *= alpha[r]; o[3][r] *= alpha[r];
      int row = r + half * 8;
      pb[row * 32 + n]      = (bf16_t)p0[r];
      pb[row * 32 + 16 + n] = (bf16_t)p1[r];
    }
    asm volatile("s_wait_dscnt 0" ::: "memory");   // same-wave LDS RAW
    v16bf ap = load_a_frag(pb, 32);                // ds_load_b128 path
    // PV: out[16x64] += P[16x32] @ V[32x64]; V^T layout makes B k-contiguous.
#pragma unroll
    for (int t = 0; t < 4; ++t) {
      v16bf bv = load_b_frag(vth + (size_t)(t * 16) * 2048 + jb, 2048);
      o[t] = wmma_bf16(ap, bv, o[t]);
    }
  }

#pragma unroll
  for (int r = 0; r < 8; ++r) {
    float inv = 1.0f / lrow[r];
    int row = i0 + r + half * 8;
    bf16_t* dst = attn + ((size_t)(b * 2048 + row)) * 1024 + h * 64;
#pragma unroll
    for (int t = 0; t < 4; ++t)
      dst[t * 16 + n] = (bf16_t)(o[t][r] * inv);
  }
}

// ---------------------------------------------------------------------------
extern "C" void kernel_launch(void* const* d_in, const int* in_sizes, int n_in,
                              void* d_out, int out_size, void* d_ws, size_t ws_size,
                              hipStream_t stream) {
  const float* x       = (const float*)d_in[0];
  const float* Wq      = (const float*)d_in[1];
  const float* Wk      = (const float*)d_in[2];
  const float* Wv      = (const float*)d_in[3];
  const float* Wo      = (const float*)d_in[4];
  const float* q_scale = (const float*)d_in[5];
  const float* k_scale = (const float*)d_in[6];
  float* out = (float*)d_out;

  const int B = 4, N = 2048, D = 1024, H = 16;
  const size_t MN = (size_t)B * N;                 // 8192 tokens

  char* ws = (char*)d_ws;
  size_t off = 0;
  auto alloc = [&](size_t bytes) {
    size_t o = off; off += (bytes + 255) & ~(size_t)255; return o;
  };
  size_t xb_o  = alloc(MN * D * sizeof(bf16_t));   // x bf16; later reused as V^T
  size_t wqb_o = alloc((size_t)D * D * sizeof(bf16_t));
  size_t wkb_o = alloc((size_t)D * D * sizeof(bf16_t));
  size_t wvb_o = alloc((size_t)D * D * sizeof(bf16_t));
  size_t wob_o = alloc((size_t)D * D * sizeof(bf16_t));
  size_t qb_o  = alloc(MN * D * sizeof(bf16_t));
  size_t kb_o  = alloc(MN * D * sizeof(bf16_t));
  size_t vb_o  = alloc(MN * D * sizeof(bf16_t));   // later reused as attn out
  size_t fq_o  = alloc((size_t)B * H * N * sizeof(float));
  size_t fk_o  = alloc((size_t)B * H * N * sizeof(float));

  bf16_t* xb  = (bf16_t*)(ws + xb_o);
  bf16_t* wqb = (bf16_t*)(ws + wqb_o);
  bf16_t* wkb = (bf16_t*)(ws + wkb_o);
  bf16_t* wvb = (bf16_t*)(ws + wvb_o);
  bf16_t* wob = (bf16_t*)(ws + wob_o);
  bf16_t* qb  = (bf16_t*)(ws + qb_o);
  bf16_t* kbuf= (bf16_t*)(ws + kb_o);
  bf16_t* vb  = (bf16_t*)(ws + vb_o);
  float*  fq  = (float*)(ws + fq_o);
  float*  fkp = (float*)(ws + fk_o);
  bf16_t* vtb  = xb;   // reuse: x bf16 dead after QKV GEMMs
  bf16_t* attn = vb;   // reuse: v bf16 dead after transpose

  // 1) convert inputs to bf16
  {
    size_t nx = MN * D;
    f32_to_bf16_kernel<<<dim3((unsigned)((nx + 255) / 256)), 256, 0, stream>>>(x, xb, nx);
    size_t nw = (size_t)D * D;
    unsigned gw = (unsigned)((nw + 255) / 256);
    f32_to_bf16_kernel<<<gw, 256, 0, stream>>>(Wq, wqb, nw);
    f32_to_bf16_kernel<<<gw, 256, 0, stream>>>(Wk, wkb, nw);
    f32_to_bf16_kernel<<<gw, 256, 0, stream>>>(Wv, wvb, nw);
    f32_to_bf16_kernel<<<gw, 256, 0, stream>>>(Wo, wob, nw);
  }

  // 2) q/k/v = silu(x @ W^T)  -> bf16
  dim3 ggrid(D / 128, (unsigned)(MN / 64));
  gemm_kernel<true,  true><<<ggrid, 128, 0, stream>>>(xb, wqb, qb,   (int)MN, D, D);
  gemm_kernel<true,  true><<<ggrid, 128, 0, stream>>>(xb, wkb, kbuf, (int)MN, D, D);
  gemm_kernel<true,  true><<<ggrid, 128, 0, stream>>>(xb, wvb, vb,   (int)MN, D, D);

  // 3) V -> V^T per head
  {
    size_t nv = MN * D;
    transpose_v_kernel<<<dim3((unsigned)((nv + 255) / 256)), 256, 0, stream>>>(vb, vtb);
  }

  // 4) scale factors
  {
    int nf = B * H * N;
    factors_kernel<<<(nf + 255) / 256, 256, 0, stream>>>(qb,   q_scale, fq,  1e-6f);
    factors_kernel<<<(nf + 255) / 256, 256, 0, stream>>>(kbuf, k_scale, fkp, 1e-6f);
  }

  // 5) causal attention
  attn_kernel<<<dim3(N / 64, B * H), 128, 0, stream>>>(qb, kbuf, vtb, fq, fkp, attn);

  // 6) final projection: out = attn @ Wo^T  (f32 out)
  gemm_kernel<false, false><<<ggrid, 128, 0, stream>>>(attn, wob, out, (int)MN, D, D);
}
